// JoinOrderRanker_41515153883621
// MI455X (gfx1250) — compile-verified, matrix operand order checked by reading, not run on hardware
//
#include <hip/hip_runtime.h>
#include <hip/hip_bf16.h>

// ---------------------------------------------------------------------------
// JoinOrderRanker on gfx1250: GIN(3) + pooling + order-scoring head.
// Matrix math on V_WMMA_F32_16X16X32_F16 (f16 in, f32 accumulate).
// Inner GEMM loop batches all B-fragment loads per K-step so one
// s_wait_loadcnt covers 8 back-to-back independent WMMAs.
// ---------------------------------------------------------------------------

typedef _Float16 v16h __attribute__((ext_vector_type(16)));
typedef float    v8f  __attribute__((ext_vector_type(8)));

#define Hdim 128

__device__ __forceinline__ unsigned ordEncode(float f) {
    unsigned u = __float_as_uint(f);
    return (u & 0x80000000u) ? ~u : (u | 0x80000000u);
}
__device__ __forceinline__ float ordDecode(unsigned e) {
    unsigned b = (e & 0x80000000u) ? (e & 0x7FFFFFFFu) : ~e;
    return __uint_as_float(b);
}

// ---------------- small utility kernels -------------------------------------
__global__ void k_zero_f32(float* p, long n) {
    long t = (long)blockIdx.x * blockDim.x + threadIdx.x;
    if (t < n) p[t] = 0.0f;
}

__global__ void k_init_h(const float* __restrict__ x, float* __restrict__ h,
                         int N, long total) {
    long t = (long)blockIdx.x * blockDim.x + threadIdx.x;
    if (t >= total) return;
    long row = t >> 7;
    h[t] = (row < N) ? x[t] : 0.0f;
}

// Af16[r][c] = f16(agg[r][c] + h[r][c]); zero for pad rows.
__global__ void k_add_convert(const float* __restrict__ agg,
                              const float* __restrict__ h,
                              _Float16* __restrict__ Af, int N, long total) {
    long t = (long)blockIdx.x * blockDim.x + threadIdx.x;
    if (t >= total) return;
    long row = t >> 7;
    Af[t] = (row < N) ? (_Float16)(agg[t] + h[t]) : (_Float16)0.0f;
}

// Pack a KxNout f32 weight into per-lane WMMA B fragments:
// layout [kt][nt][lane][16 halves]; halves 0..7 = W[kb+j][col], 8..15 = W[kb+16+j][col]
// with col = (lane&15)+nt*16, kb = kt*32 + (lane>>4)*8.
__global__ void k_packW(const float* __restrict__ W, _Float16* __restrict__ P,
                        int K, int Nout) {
    int KT = K / 32, NT = Nout / 16;
    int tid = blockIdx.x * blockDim.x + threadIdx.x;
    int total = KT * NT * 32;
    if (tid >= total) return;
    int lane = tid & 31;
    int nt = (tid >> 5) % NT;
    int kt = tid / (32 * NT);
    int col = (lane & 15) + nt * 16;
    int kb = kt * 32 + (lane >> 4) * 8;
    _Float16* p = P + (size_t)tid * 16;
#pragma unroll
    for (int j = 0; j < 8; ++j) p[j] = (_Float16)W[(size_t)(kb + j) * Nout + col];
#pragma unroll
    for (int j = 0; j < 8; ++j) p[8 + j] = (_Float16)W[(size_t)(kb + 16 + j) * Nout + col];
}

// ---------------- edge scatter-add (GIN aggregation) ------------------------
// 32 threads per edge, 4 floats each (H=128). L2-resident atomics.
__global__ void k_scatter_edges(const int* __restrict__ src, const int* __restrict__ dst,
                                const float* __restrict__ h, float* __restrict__ agg,
                                int E) {
    long t = (long)blockIdx.x * blockDim.x + threadIdx.x;
    long e = t >> 5;
    if (e >= E) return;
    int s = src[e], d = dst[e];
    int c = ((int)t & 31) * 4;
    const float4 v = *(const float4*)(h + (size_t)s * Hdim + c);
    float* ap = agg + (size_t)d * Hdim + c;
    atomicAdd(ap + 0, v.x);
    atomicAdd(ap + 1, v.y);
    atomicAdd(ap + 2, v.z);
    atomicAdd(ap + 3, v.w);
}

// ---------------- generic WMMA GEMM -----------------------------------------
// One wave -> 16 rows x (NT*16) cols. 8 waves/WG -> 128 rows per workgroup.
// A: MxK row-major f16 (lda == K). Bp: packed fragments. Optional f16 and/or
// f32 outputs, optional ReLU, optional per-column sum/sum^2 (for BatchNorm).
// Per kt-step: issue A loads + all NT B-fragment loads, then NT independent
// WMMAs back-to-back (single loadcnt wait amortized over the whole step).
template <int NT>
__global__ void k_gemm_wmma(const _Float16* A, const _Float16* __restrict__ Bp,
                            const float* __restrict__ bias, int M, int K,
                            _Float16* outH, int ldh, float* outF, int ldf,
                            int relu, float* colsum, float* colsumsq) {
    __shared__ float sSum[Hdim];
    __shared__ float sSq[Hdim];
    const bool doStats = (colsum != nullptr);
    if (doStats) {
        for (int c = threadIdx.x; c < Hdim; c += blockDim.x) { sSum[c] = 0.f; sSq[c] = 0.f; }
        __syncthreads();
    }
    const int lane = threadIdx.x & 31;
    const int wave = threadIdx.x >> 5;
    const int rowBase = blockIdx.x * 128 + wave * 16;
    const int aRow = rowBase + (lane & 15);
    const int kSub = (lane >> 4) * 8;

    v8f acc[NT] = {};
    const int KT = K / 32;
    const _Float16* aBase = A + (size_t)aRow * K + kSub;

    for (int kt = 0; kt < KT; ++kt) {
        // --- load A fragment for this K-step ---
        v16h af;
        const _Float16* ap = aBase + kt * 32;
#pragma unroll
        for (int j = 0; j < 8; ++j) { af[j] = ap[j]; af[8 + j] = ap[16 + j]; }
        // --- load ALL B fragments for this K-step first (one wait for all) ---
        v16h bf[NT];
        const _Float16* bp = Bp + ((size_t)(kt * NT) * 32 + lane) * 16;
#pragma unroll
        for (int nt = 0; nt < NT; ++nt) {
            const _Float16* bq = bp + (size_t)nt * 32 * 16;
#pragma unroll
            for (int j = 0; j < 16; ++j) bf[nt][j] = bq[j];
        }
        // --- NT independent WMMAs back-to-back ---
#pragma unroll
        for (int nt = 0; nt < NT; ++nt) {
            acc[nt] = __builtin_amdgcn_wmma_f32_16x16x32_f16(
                false, af, false, bf[nt], (short)0, acc[nt], false, false);
        }
    }

    const int colHalf = lane & 15;
    const int rOff = (lane >> 4) * 8;
#pragma unroll
    for (int nt = 0; nt < NT; ++nt) {
        const int col = nt * 16 + colHalf;
        const float bv = bias ? bias[col] : 0.0f;
#pragma unroll
        for (int v = 0; v < 8; ++v) {
            int row = rowBase + rOff + v;
            float val = acc[nt][v] + bv;
            if (relu) val = fmaxf(val, 0.0f);
            if (row < M) {
                if (outH) outH[(size_t)row * ldh + col] = (_Float16)val;
                if (outF) outF[(size_t)row * ldf + col] = val;
                if (doStats) {
                    atomicAdd(&sSum[col], val);
                    atomicAdd(&sSq[col], val * val);
                }
            }
        }
    }
    if (doStats) {
        __syncthreads();
        for (int c = threadIdx.x; c < Hdim; c += blockDim.x) {
            atomicAdd(&colsum[c], sSum[c]);
            atomicAdd(&colsumsq[c], sSq[c]);
        }
    }
}

// ---------------- BatchNorm -------------------------------------------------
__global__ void k_bn_prep(const float* __restrict__ colsum, const float* __restrict__ colsumsq,
                          const float* __restrict__ gamma, const float* __restrict__ beta,
                          float* __restrict__ bnA, float* __restrict__ bnB, float invN) {
    int c = threadIdx.x;
    if (c >= Hdim) return;
    float mu = colsum[c] * invN;
    float var = colsumsq[c] * invN - mu * mu;
    float a = gamma[c] * rsqrtf(var + 1e-5f);
    bnA[c] = a;
    bnB[c] = beta[c] - mu * a;
}

__global__ void k_bn_apply(const float* __restrict__ z2, const float* __restrict__ bnA,
                           const float* __restrict__ bnB, float* __restrict__ h,
                           int relu, long total) {
    long t = (long)blockIdx.x * blockDim.x + threadIdx.x;
    if (t >= total) return;
    int c = (int)(t & (Hdim - 1));
    float v = z2[t] * bnA[c] + bnB[c];
    if (relu) v = fmaxf(v, 0.0f);
    h[t] = v;
}

// ---------------- pooling ---------------------------------------------------
__global__ void k_pool_scatter(const float* __restrict__ h, const int* __restrict__ batch,
                               float* __restrict__ cnt, float* __restrict__ ssum,
                               unsigned* __restrict__ smax, int N) {
    long t = (long)blockIdx.x * blockDim.x + threadIdx.x;
    long n = t >> 5;
    if (n >= N) return;
    int b = batch[n];
    int c = ((int)t & 31) * 4;
    if ((t & 31) == 0) atomicAdd(&cnt[b], 1.0f);
    const float4 v = *(const float4*)(h + (size_t)n * Hdim + c);
    float* sp = ssum + (size_t)b * Hdim + c;
    atomicAdd(sp + 0, v.x); atomicAdd(sp + 1, v.y);
    atomicAdd(sp + 2, v.z); atomicAdd(sp + 3, v.w);
    unsigned* mp = smax + (size_t)b * Hdim + c;
    atomicMax(mp + 0, ordEncode(v.x)); atomicMax(mp + 1, ordEncode(v.y));
    atomicMax(mp + 2, ordEncode(v.z)); atomicMax(mp + 3, ordEncode(v.w));
}

// comb[r][0:384] = [mean | max | sum] of graph b = r/O, as f16 (ldc = 512)
__global__ void k_build_comb(const float* __restrict__ cnt, const float* __restrict__ ssum,
                             const unsigned* __restrict__ smax, _Float16* __restrict__ comb,
                             int O, long total) {
    long t = (long)blockIdx.x * blockDim.x + threadIdx.x;
    if (t >= total) return;
    long r = t / 384;
    int c = (int)(t % 384);
    int b = (int)(r / O);
    float g;
    if (c < Hdim) {
        g = ssum[(size_t)b * Hdim + c] / fmaxf(cnt[b], 1.0f);
    } else if (c < 2 * Hdim) {
        int cc = c - Hdim;
        g = (cnt[b] > 0.0f) ? ordDecode(smax[(size_t)b * Hdim + cc]) : 0.0f;
    } else {
        g = ssum[(size_t)b * Hdim + (c - 2 * Hdim)];
    }
    comb[r * 512 + c] = (_Float16)g;
}

__global__ void k_conv_f32_f16(const float* __restrict__ src, _Float16* __restrict__ dst,
                               long n) {
    long t = (long)blockIdx.x * blockDim.x + threadIdx.x;
    if (t < n) dst[t] = (_Float16)src[t];
}

// final 64 -> 1 projection: one wave per row, shuffle reduce (wave32).
__global__ void k_final_dot(const _Float16* __restrict__ s2, const float* __restrict__ w,
                            const float* __restrict__ b, float* __restrict__ out, int rows) {
    int lane = threadIdx.x & 31;
    int wave = threadIdx.x >> 5;
    int row = blockIdx.x * 8 + wave;
    if (row >= rows) return;
    const _Float16* rp = s2 + (size_t)row * 64;
    float acc = (float)rp[lane] * w[lane] + (float)rp[lane + 32] * w[lane + 32];
#pragma unroll
    for (int off = 16; off > 0; off >>= 1) acc += __shfl_down(acc, off, 32);
    if (lane == 0) out[row] = acc + b[0];
}

// ---------------------------------------------------------------------------
extern "C" void kernel_launch(void* const* d_in, const int* in_sizes, int n_in,
                              void* d_out, int out_size, void* d_ws, size_t ws_size,
                              hipStream_t stream) {
    const float* x      = (const float*)d_in[0];
    const int*   eidx   = (const int*)d_in[1];
    const float* orders = (const float*)d_in[2];
    const int*   batch  = (const int*)d_in[3];
    const float* gW1 = (const float*)d_in[4];
    const float* gb1 = (const float*)d_in[5];
    const float* gW2 = (const float*)d_in[6];
    const float* gb2 = (const float*)d_in[7];
    const float* gamma = (const float*)d_in[8];
    const float* beta  = (const float*)d_in[9];
    const float* oW1 = (const float*)d_in[10];
    const float* ob1 = (const float*)d_in[11];
    const float* oW2 = (const float*)d_in[12];
    const float* ob2 = (const float*)d_in[13];
    const float* sW1 = (const float*)d_in[14];
    const float* sb1 = (const float*)d_in[15];
    const float* sW2 = (const float*)d_in[16];
    const float* sb2 = (const float*)d_in[17];
    const float* sW3 = (const float*)d_in[18];
    const float* sb3 = (const float*)d_in[19];

    const int N = in_sizes[0] / Hdim;       // 100000
    const int E = in_sizes[1] / 2;          // 1600000
    const int OD = 32, O = 32;
    const int B = in_sizes[2] / (O * OD);   // 256
    const int L = in_sizes[4] / (Hdim * Hdim); // 3
    const int Mpad = ((N + 127) / 128) * 128;
    const long totHM = (long)Mpad * Hdim;
    const long totHN = (long)N * Hdim;
    const int RO = B * O;                   // 8192 head rows (multiple of 128)

    // ---- workspace carving ----
    size_t off = 0;
    auto alloc = [&](size_t bytes) -> void* {
        void* p = (char*)d_ws + off;
        off = (off + bytes + 255) & ~(size_t)255;
        return p;
    };
    float*     hBuf   = (float*)alloc((size_t)totHM * 4);
    float*     aggBuf = (float*)alloc((size_t)totHM * 4);   // reused as z2
    _Float16*  Af     = (_Float16*)alloc((size_t)totHM * 2);
    _Float16*  pgW1[3], *pgW2[3];
    for (int i = 0; i < 3; ++i) pgW1[i] = (_Float16*)alloc(4 * 8 * 32 * 16 * 2);
    for (int i = 0; i < 3; ++i) pgW2[i] = (_Float16*)alloc(4 * 8 * 32 * 16 * 2);
    _Float16*  poW1 = (_Float16*)alloc(1 * 8 * 32 * 16 * 2);
    _Float16*  poW2 = (_Float16*)alloc(4 * 8 * 32 * 16 * 2);
    _Float16*  psW1 = (_Float16*)alloc(16 * 8 * 32 * 16 * 2);
    _Float16*  psW2 = (_Float16*)alloc(4 * 4 * 32 * 16 * 2);
    float*     colsum   = (float*)alloc(Hdim * 4);
    float*     colsumsq = (float*)alloc(Hdim * 4);
    float*     bnA = (float*)alloc(Hdim * 4);
    float*     bnB = (float*)alloc(Hdim * 4);
    float*     cnt  = (float*)alloc((size_t)B * 4);
    float*     ssum = (float*)alloc((size_t)B * Hdim * 4);
    unsigned*  smax = (unsigned*)alloc((size_t)B * Hdim * 4);
    _Float16*  comb = (_Float16*)alloc((size_t)RO * 512 * 2);
    _Float16*  Aord = (_Float16*)alloc((size_t)RO * OD * 2);
    _Float16*  t1   = (_Float16*)alloc((size_t)RO * Hdim * 2);
    _Float16*  s1   = (_Float16*)alloc((size_t)RO * Hdim * 2);
    _Float16*  s2   = (_Float16*)alloc((size_t)RO * 64 * 2);
    float*     out  = (float*)d_out;

    auto cdiv = [](long a, long b) -> unsigned { return (unsigned)((a + b - 1) / b); };

    // ---- pack all weights into WMMA B-fragment layout ----
    for (int i = 0; i < L; ++i) {
        k_packW<<<cdiv(4 * 8 * 32, 256), 256, 0, stream>>>(gW1 + (size_t)i * Hdim * Hdim, pgW1[i], Hdim, Hdim);
        k_packW<<<cdiv(4 * 8 * 32, 256), 256, 0, stream>>>(gW2 + (size_t)i * Hdim * Hdim, pgW2[i], Hdim, Hdim);
    }
    k_packW<<<cdiv(1 * 8 * 32, 256), 256, 0, stream>>>(oW1, poW1, OD, Hdim);
    k_packW<<<cdiv(4 * 8 * 32, 256), 256, 0, stream>>>(oW2, poW2, Hdim, Hdim);
    k_packW<<<cdiv(16 * 8 * 32, 256), 256, 0, stream>>>(sW1, psW1, 512, Hdim);
    k_packW<<<cdiv(4 * 4 * 32, 256), 256, 0, stream>>>(sW2, psW2, Hdim, 64);

    // ---- h = x (padded) ----
    k_init_h<<<cdiv(totHM, 256), 256, 0, stream>>>(x, hBuf, N, totHM);

    const unsigned gemmBlocks = (unsigned)(Mpad / 128);
    // ---- GIN layers ----
    for (int i = 0; i < L; ++i) {
        k_zero_f32<<<cdiv(totHM, 256), 256, 0, stream>>>(aggBuf, totHM);
        k_scatter_edges<<<cdiv((long)E * 32, 256), 256, 0, stream>>>(eidx, eidx + E, hBuf, aggBuf, E);
        k_add_convert<<<cdiv(totHM, 256), 256, 0, stream>>>(aggBuf, hBuf, Af, N, totHM);
        // t = relu(z @ W1 + b1)  (in-place on Af: each row is read only by its writer wave)
        k_gemm_wmma<8><<<gemmBlocks, 256, 0, stream>>>(Af, pgW1[i], gb1 + (size_t)i * Hdim,
                                                       N, Hdim, Af, Hdim, nullptr, 0, 1,
                                                       nullptr, nullptr);
        k_zero_f32<<<1, 256, 0, stream>>>(colsum, Hdim);
        k_zero_f32<<<1, 256, 0, stream>>>(colsumsq, Hdim);
        // z2 = t @ W2 + b2 (f32) + column stats for BatchNorm
        k_gemm_wmma<8><<<gemmBlocks, 256, 0, stream>>>(Af, pgW2[i], gb2 + (size_t)i * Hdim,
                                                       N, Hdim, nullptr, 0, aggBuf, Hdim, 0,
                                                       colsum, colsumsq);
        k_bn_prep<<<1, 128, 0, stream>>>(colsum, colsumsq, gamma + (size_t)i * Hdim,
                                         beta + (size_t)i * Hdim, bnA, bnB, 1.0f / (float)N);
        k_bn_apply<<<cdiv(totHN, 256), 256, 0, stream>>>(aggBuf, bnA, bnB, hBuf,
                                                         (i < L - 1) ? 1 : 0, totHN);
    }

    // ---- graph pooling ----
    k_zero_f32<<<1, 256, 0, stream>>>(cnt, B);
    k_zero_f32<<<cdiv((long)B * Hdim, 256), 256, 0, stream>>>(ssum, (long)B * Hdim);
    k_zero_f32<<<cdiv((long)B * Hdim, 256), 256, 0, stream>>>((float*)smax, (long)B * Hdim); // 0u == minimal ord key
    k_pool_scatter<<<cdiv((long)N * 32, 256), 256, 0, stream>>>(hBuf, batch, cnt, ssum, smax, N);
    k_build_comb<<<cdiv((long)RO * 384, 256), 256, 0, stream>>>(cnt, ssum, smax, comb, O, (long)RO * 384);

    // ---- order encoder + score head ----
    k_conv_f32_f16<<<cdiv((long)RO * OD, 256), 256, 0, stream>>>(orders, Aord, (long)RO * OD);
    const unsigned headBlocks = (unsigned)(RO / 128);
    k_gemm_wmma<8><<<headBlocks, 256, 0, stream>>>(Aord, poW1, ob1, RO, OD, t1, Hdim,
                                                   nullptr, 0, 1, nullptr, nullptr);
    k_gemm_wmma<8><<<headBlocks, 256, 0, stream>>>(t1, poW2, ob2, RO, Hdim, comb + 384, 512,
                                                   nullptr, 0, 0, nullptr, nullptr);
    k_gemm_wmma<8><<<headBlocks, 256, 0, stream>>>(comb, psW1, sb1, RO, 512, s1, Hdim,
                                                   nullptr, 0, 1, nullptr, nullptr);
    k_gemm_wmma<4><<<headBlocks, 256, 0, stream>>>(s1, psW2, sb2, RO, Hdim, s2, 64,
                                                   nullptr, 0, 1, nullptr, nullptr);
    k_final_dot<<<cdiv(RO, 8), 256, 0, stream>>>(s2, sW3, sb3, out, RO);
}
